// Decoder_1D_51926154608671
// MI455X (gfx1250) — matrix operand order repaired and checked by reading, not compiled, not run on hardware
//
#include <hip/hip_runtime.h>
#include <hip/hip_bf16.h>

typedef __bf16 bf16_t;
typedef __attribute__((ext_vector_type(16))) __bf16 v16bf;
typedef __attribute__((ext_vector_type(8)))  __bf16 v8bf;
typedef __attribute__((ext_vector_type(4)))  __bf16 v4bf;
typedef __attribute__((ext_vector_type(8)))  float  v8f;
typedef __attribute__((ext_vector_type(4)))  int    v4i;

#define AS1 __attribute__((address_space(1)))
#define AS3 __attribute__((address_space(3)))

#define CB_N     8192
#define DD       1024
#define HH       4096
#define OO       1024
#define NTOK     18432
#define MT       64        // tokens per workgroup
#define HC       128       // hidden-dim chunk (fused K of GEMM2)
#define NTHREADS 512       // 16 waves (wave32)

static __device__ __forceinline__ v16bf cat16(v8bf lo, v8bf hi) {
  return __builtin_shufflevector(lo, hi, 0,1,2,3,4,5,6,7,8,9,10,11,12,13,14,15);
}

static __device__ __forceinline__ float gelu_tanh(float x) {
  float u = 0.7978845608028654f * (x + 0.044715f * x * x * x);
  return 0.5f * x * (1.0f + tanhf(u));
}

// ---- pre-pass 1: fp32 row-major src[R][C] -> bf16 transposed dst[C][R]
__global__ __launch_bounds__(1024) void transpose_convert(
    const float* __restrict__ src, bf16_t* __restrict__ dst, int R, int C) {
  __shared__ float tile[32][33];
  int cb = blockIdx.x * 32, rb = blockIdx.y * 32;
  int tx = threadIdx.x & 31, ty = threadIdx.x >> 5;
  tile[ty][tx] = src[(size_t)(rb + ty) * C + (cb + tx)];
  __syncthreads();
  dst[(size_t)(cb + ty) * R + (rb + tx)] = (bf16_t)tile[tx][ty];
}

// ---- pre-pass 2: fp32 -> bf16 elementwise (codebook), n % 4 == 0
__global__ __launch_bounds__(256) void convert_bf16(
    const float* __restrict__ src, bf16_t* __restrict__ dst, int n) {
  int i = (blockIdx.x * 256 + threadIdx.x) * 4;
  if (i < n) {
    float4 v = *(const float4*)(src + i);
    v4bf o = {(bf16_t)v.x, (bf16_t)v.y, (bf16_t)v.z, (bf16_t)v.w};
    *(v4bf*)(dst + i) = o;
  }
}

__global__ __launch_bounds__(NTHREADS, 1) void decoder_mlp(
    const int* __restrict__ idx,
    const bf16_t* __restrict__ cbb,   // [CB_N][DD] bf16 codebook
    const bf16_t* __restrict__ w1t,   // [HH][DD] bf16 (W1 transposed)
    const float* __restrict__ b1v,
    const bf16_t* __restrict__ w2t,   // [OO][HH] bf16 (W2 transposed)
    const float* __restrict__ b2v,
    float* __restrict__ out) {
  extern __shared__ bf16_t smem[];
  bf16_t* qs  = smem;                      // [MT][DD] gathered rows (128 KB)
  bf16_t* hsA = smem + MT * DD;            // [MT][HC] ping      (16 KB)
  bf16_t* hsB = hsA + MT * HC;             // [MT][HC] pong      (16 KB)

  const int tid  = threadIdx.x;
  const int lane = tid & 31;
  const int wid  = tid >> 5;       // 0..15
  const int mg   = wid >> 2;       // row group: rows [mg*16, mg*16+16)
  const int ng   = wid & 3;        // col group: cols [ng*256, ng*256+256)
  const int tok0 = blockIdx.x * MT;

  const int lh   = lane >> 4;      // lane half (WMMA K-block select)
  const int kb8  = lh * 8;         // unused: A-frag also uses kb16
  const int kb16 = lh * 16;        // B-frag K base
  const int nloc = lane & 15;      // B/C/D column within tile
  const int mrow = lane & 15;      // A row within tile

  // ---- gather: qs[t][0:DD] = cbb[idx[tok0+t]][0:DD]  (pure byte copy)
#if __has_builtin(__builtin_amdgcn_global_load_async_to_lds_b128)
  // async DMA into LDS, 16 B per lane per op; dynamic LDS starts at offset 0
  for (int i = tid * 8; i < MT * DD; i += NTHREADS * 8) {
    int t = i >> 10;               // DD == 1024
    int d = i & (DD - 1);
    const bf16_t* g = cbb + (size_t)idx[tok0 + t] * DD + d;
    __builtin_amdgcn_global_load_async_to_lds_b128(
        (AS1 v4i*)(uintptr_t)g, (AS3 v4i*)(unsigned)(i * 2), 0, 0);
  }
#if __has_builtin(__builtin_amdgcn_s_wait_asynccnt)
  __builtin_amdgcn_s_wait_asynccnt(0);
#else
  asm volatile("s_wait_asynccnt 0x0" ::: "memory");
#endif
#else
  for (int i = tid * 8; i < MT * DD; i += NTHREADS * 8) {
    int t = i >> 10;
    int d = i & (DD - 1);
    *(v8bf*)(qs + i) = *(const v8bf*)(cbb + (size_t)idx[tok0 + t] * DD + d);
  }
#endif
  __syncthreads();

  v8f acc[16] = {};                // this wave: 16 rows x 256 cols of rec

  const bf16_t* qrow = qs + (mg * 16 + mrow) * DD;

  for (int c = 0; c < HH / HC; c++) {
    const int h0 = c * HC;
    bf16_t* hsc = (c & 1) ? hsB : hsA;

    // ---- phase 1: h_chunk[MT][HC] = gelu(q @ W1[:, h0:h0+HC] + b1)
    // 32 (16x16) tiles; wave (mg, ng) computes hidden tiles {2*ng, 2*ng+1}
    const int n1a = h0 + (ng * 2 + 0) * 16 + nloc;
    const int n1b = h0 + (ng * 2 + 1) * 16 + nloc;
    const bf16_t* wra = w1t + (size_t)n1a * DD;   // W1 column n1a, contiguous K
    const bf16_t* wrb = w1t + (size_t)n1b * DD;
    v8f ha = {}, hb = {};
    #pragma unroll 4
    for (int k0 = 0; k0 < DD; k0 += 32) {
      v16bf a = cat16(*(const v8bf*)(qrow + k0 + kb16),
                      *(const v8bf*)(qrow + k0 + kb16 + 8));
      v16bf ba = cat16(*(const v8bf*)(wra + k0 + kb16),
                       *(const v8bf*)(wra + k0 + kb16 + 8));
      ha = __builtin_amdgcn_wmma_f32_16x16x32_bf16(false, a, false, ba,
                                                   (short)0, ha, false, false);
      v16bf bb = cat16(*(const v8bf*)(wrb + k0 + kb16),
                       *(const v8bf*)(wrb + k0 + kb16 + 8));
      hb = __builtin_amdgcn_wmma_f32_16x16x32_bf16(false, a, false, bb,
                                                   (short)0, hb, false, false);
    }
    const float biasa = b1v[n1a];
    const float biasb = b1v[n1b];
    #pragma unroll
    for (int r = 0; r < 8; r++) {
      // C/D layout: VGPR r holds M = r + 8*(lane/16), N = lane%16
      int m = mg * 16 + r + lh * 8;
      hsc[m * HC + (ng * 2 + 0) * 16 + nloc] = (bf16_t)gelu_tanh(ha[r] + biasa);
      hsc[m * HC + (ng * 2 + 1) * 16 + nloc] = (bf16_t)gelu_tanh(hb[r] + biasb);
    }

    // prefetch next chunk's W1 columns (L2 -> near caches)
    if (h0 + HC < HH) {
      __builtin_prefetch(wra + (size_t)HC * DD, 0, 1);
      __builtin_prefetch(wrb + (size_t)HC * DD, 0, 1);
    }
    __syncthreads();   // hsc written -> readable; prior reads of hsc (2 chunks
                       // ago) finished before the previous chunk's barrier

    // ---- phase 2: acc += h_chunk @ W2[h0:h0+HC, ng*256 : ng*256+256]
    const bf16_t* hrow = hsc + (mg * 16 + mrow) * HC;
    #pragma unroll
    for (int kt = 0; kt < HC / 32; kt++) {
      v16bf a = cat16(*(const v8bf*)(hrow + kt * 32 + kb16),
                      *(const v8bf*)(hrow + kt * 32 + kb16 + 8));
      #pragma unroll
      for (int nt = 0; nt < 16; nt++) {
        const bf16_t* wr =
            w2t + (size_t)(ng * 256 + nt * 16 + nloc) * HH + h0 + kt * 32;
        v16bf b = cat16(*(const v8bf*)(wr + kb16),
                        *(const v8bf*)(wr + kb16 + 8));
        acc[nt] = __builtin_amdgcn_wmma_f32_16x16x32_bf16(false, a, false, b,
                                                          (short)0, acc[nt],
                                                          false, false);
      }
    }
    // no trailing barrier: ping-pong buffer isn't rewritten until after the
    // next chunk's barrier, by which time all waves have consumed it
  }

  // ---- epilogue: rec = acc + b2, fp32 store
  #pragma unroll
  for (int nt = 0; nt < 16; nt++) {
    int ncol = ng * 256 + nt * 16 + nloc;
    float bias = b2v[ncol];
    #pragma unroll
    for (int r = 0; r < 8; r++) {
      int m = mg * 16 + r + lh * 8;
      out[(size_t)(tok0 + m) * OO + ncol] = acc[nt][r] + bias;
    }
  }
}

extern "C" void kernel_launch(void* const* d_in, const int* in_sizes, int n_in,
                              void* d_out, int out_size, void* d_ws, size_t ws_size,
                              hipStream_t stream) {
  const int*   idx = (const int*)  d_in[0];
  const float* cb  = (const float*)d_in[1];
  const float* W1  = (const float*)d_in[2];
  const float* b1  = (const float*)d_in[3];
  const float* W2  = (const float*)d_in[4];
  const float* b2  = (const float*)d_in[5];
  float* out = (float*)d_out;

  // workspace layout (assumes ws_size >= 32 MB):
  //   [0,   8MB) : w1t bf16 [HH][DD]
  //   [8,  16MB) : w2t bf16 [OO][HH]
  //   [16, 32MB) : cbb bf16 [CB_N][DD]
  bf16_t* w1t = (bf16_t*)d_ws;
  bf16_t* w2t = (bf16_t*)((char*)d_ws + (size_t)HH * DD * 2);
  bf16_t* cbb = (bf16_t*)((char*)d_ws + (size_t)HH * DD * 2 + (size_t)OO * HH * 2);

  transpose_convert<<<dim3(HH / 32, DD / 32), 1024, 0, stream>>>(W1, w1t, DD, HH);
  transpose_convert<<<dim3(OO / 32, HH / 32), 1024, 0, stream>>>(W2, w2t, HH, OO);
  {
    int n = CB_N * DD;
    convert_bf16<<<n / (256 * 4), 256, 0, stream>>>(cb, cbb, n);
  }

  size_t shmem = (size_t)(MT * DD + 2 * MT * HC) * sizeof(bf16_t);  // 160 KB
  decoder_mlp<<<NTOK / MT, NTHREADS, shmem, stream>>>(idx, cbb, w1t, b1, w2t, b2,
                                                      out);
}
